// BiLSTM_10411000725813
// MI455X (gfx1250) — matrix operand order repaired and checked by reading, not compiled, not run on hardware
//
#include <hip/hip_runtime.h>

#define BB 16
#define LL 2048
#define EE 512
#define UU 512
#define GG 2048  // 4*UU

typedef __attribute__((ext_vector_type(16))) __bf16 v16bf;
typedef __attribute__((ext_vector_type(8)))  __bf16 v8bf;
typedef __attribute__((ext_vector_type(8)))  float  v8f;

static __device__ __forceinline__ v16bf concat16(v8bf lo, v8bf hi) {
  return __builtin_shufflevector(lo, hi, 0,1,2,3,4,5,6,7,8,9,10,11,12,13,14,15);
}
static __device__ __forceinline__ float fsigm(float x) {
  return 1.0f / (1.0f + __expf(-x));
}
static __device__ __forceinline__ float ftanh(float x) {
  return 2.0f / (1.0f + __expf(-2.0f * x)) - 1.0f;
}

// ---------------------------------------------------------------------------
// x [B,L,E] f32  ->  Xbf [(t*B+b), e] bf16  (time-major rows for GEMM/recurrence)
// ---------------------------------------------------------------------------
__global__ __launch_bounds__(256) void cvt_x_kernel(const float* __restrict__ x,
                                                    __bf16* __restrict__ Xbf) {
  size_t idx = (size_t)blockIdx.x * 256 + threadIdx.x;  // over B*L*E, dest order
  int e = (int)(idx % EE);
  size_t r = idx / EE;        // t*B + b
  int b = (int)(r & 15);
  size_t t = r >> 4;
  Xbf[idx] = (__bf16)x[((size_t)b * LL + t) * EE + e];
}

// ---------------------------------------------------------------------------
// Pack f32 row-major weight [K,N] into bf16 WMMA B-fragment order:
// frag id = nt*(K/32)+kt ; within frag: lane-major, lane holds 16 (two runs of 8)
// K values for its column:  n = nt*16 + (lane&15), k = kt*32 + (lane>=16?8:0)+e+(e>=8?8:0)
// ---------------------------------------------------------------------------
__global__ __launch_bounds__(256) void pack_b_kernel(const float* __restrict__ src,
                                                     __bf16* __restrict__ dst,
                                                     int K, int N) {
  int id = blockIdx.x * 256 + threadIdx.x;  // frag*32 + lane
  int lane = id & 31;
  int frag = id >> 5;
  int kfrags = K >> 5;
  int nt = frag / kfrags;
  if (nt >= (N >> 4)) return;
  int kt = frag - nt * kfrags;
  int n = nt * 16 + (lane & 15);
  int kbase = kt * 32 + ((lane >> 4) << 3);
  __bf16* d = dst + (size_t)frag * 512 + lane * 16;
#pragma unroll
  for (int e = 0; e < 16; ++e) d[e] = (__bf16)src[(size_t)(kbase + e + (e >= 8 ? 8 : 0)) * N + n];
}

// ---------------------------------------------------------------------------
// WMMA GEMM: C = A(bf16 row-major [M,K]) * Bp(packed bf16) + bias
// mode 0: store bf16 Z interleaved  [t][N][16]  (t = m>>4, b = m&15)
// mode 1: store f32 out             [b][t][n]   (projection -> d_out [B,L,U])
// Block: 256 thr (8 waves), tile 128M x 64N, wave tile 32x32 (2x2 subtiles).
// ---------------------------------------------------------------------------
__global__ __launch_bounds__(256) void gemm_wmma_kernel(
    const __bf16* __restrict__ A, const __bf16* __restrict__ Bp,
    const float* __restrict__ bias, void* __restrict__ Cout,
    int M, int N, int K, int mode) {
  const int lane = threadIdx.x & 31;
  const int wave = threadIdx.x >> 5;
  const int wm = wave >> 1, wn = wave & 1;
  const int m0 = blockIdx.x * 128 + wm * 32;
  const int n0 = blockIdx.y * 64 + wn * 32;
  const int l15 = lane & 15;
  const int lh8 = (lane >> 4) << 3;  // 0 or 8
  const int kfrags = K >> 5;
  (void)M;

  v8f acc[2][2];
#pragma unroll
  for (int i = 0; i < 2; ++i)
#pragma unroll
    for (int j = 0; j < 2; ++j) {
      float bv = bias[n0 + j * 16 + l15];
#pragma unroll
      for (int v = 0; v < 8; ++v) acc[i][j][v] = bv;
    }

  const __bf16* Arow0 = A + (size_t)(m0 + l15) * K;
  const __bf16* Arow1 = Arow0 + (size_t)16 * K;
  const __bf16* Bf0 = Bp + (size_t)(n0 >> 4) * kfrags * 512 + lane * 16;
  const __bf16* Bf1 = Bf0 + (size_t)kfrags * 512;

  for (int kt = 0; kt < kfrags; ++kt) {
    const int kb = kt * 32 + lh8;
    v16bf a0 = concat16(*(const v8bf*)(Arow0 + kb), *(const v8bf*)(Arow0 + kb + 16));
    v16bf a1 = concat16(*(const v8bf*)(Arow1 + kb), *(const v8bf*)(Arow1 + kb + 16));
    const __bf16* p0 = Bf0 + (size_t)kt * 512;
    const __bf16* p1 = Bf1 + (size_t)kt * 512;
    v16bf b0 = concat16(*(const v8bf*)p0, *(const v8bf*)(p0 + 8));
    v16bf b1 = concat16(*(const v8bf*)p1, *(const v8bf*)(p1 + 8));
    acc[0][0] = __builtin_amdgcn_wmma_f32_16x16x32_bf16(false, a0, false, b0, (short)0, acc[0][0], false, false);
    acc[0][1] = __builtin_amdgcn_wmma_f32_16x16x32_bf16(false, a0, false, b1, (short)0, acc[0][1], false, false);
    acc[1][0] = __builtin_amdgcn_wmma_f32_16x16x32_bf16(false, a1, false, b0, (short)0, acc[1][0], false, false);
    acc[1][1] = __builtin_amdgcn_wmma_f32_16x16x32_bf16(false, a1, false, b1, (short)0, acc[1][1], false, false);
  }

  if (mode == 0) {
    __bf16* Z = (__bf16*)Cout;
#pragma unroll
    for (int i = 0; i < 2; ++i) {
      size_t t = (size_t)((m0 + i * 16) >> 4);
#pragma unroll
      for (int j = 0; j < 2; ++j) {
        int n = n0 + j * 16 + l15;
        v8bf outv = __builtin_convertvector(acc[i][j], v8bf);
        *(v8bf*)(Z + t * ((size_t)N * 16) + (size_t)n * 16 + lh8) = outv;
      }
    }
  } else {
    float* Out = (float*)Cout;
#pragma unroll
    for (int i = 0; i < 2; ++i) {
      size_t t = (size_t)((m0 + i * 16) >> 4);
#pragma unroll
      for (int j = 0; j < 2; ++j) {
        int n = n0 + j * 16 + l15;
#pragma unroll
        for (int v = 0; v < 8; ++v) {
          size_t b = (size_t)(lh8 + v);
          Out[b * ((size_t)LL * UU) + t * UU + n] = acc[i][j][v];
        }
      }
    }
  }
}

// ---------------------------------------------------------------------------
// Recurrent LSTM step kernel. grid.x = 2 (0=fwd, 1=bwd), block = 1024 (32 waves).
// h_{t-1} in LDS (16x512 bf16). Wave w owns hidden cols [16w,16w+16);
// per step: 4 gate tiles x 16 K-frags of v_wmma_f32_16x16x32_bf16,
// c kept in f32 accumulator registers, gates fused elementwise.
// ---------------------------------------------------------------------------
__global__ __launch_bounds__(1024) void lstm_rec_kernel(
    const __bf16* __restrict__ Zf, const __bf16* __restrict__ Zb,
    const __bf16* __restrict__ Upf, const __bf16* __restrict__ Upb,
    __bf16* __restrict__ Hf, __bf16* __restrict__ Hb,
    int ldh, int coff_f, int coff_b) {
  __shared__ __align__(16) __bf16 hbuf[16 * 512];
  const int dir = blockIdx.x;
  const __bf16* Z  = dir ? Zb : Zf;
  const __bf16* Up = dir ? Upb : Upf;
  __bf16* H        = dir ? Hb : Hf;
  const int coff   = dir ? coff_b : coff_f;

  const int lane = threadIdx.x & 31;
  const int w    = threadIdx.x >> 5;  // 0..31
  const int n0   = w * 16;
  const int l15  = lane & 15;
  const int lh8  = (lane >> 4) << 3;

  for (int i = threadIdx.x; i < 16 * 512; i += 1024) hbuf[i] = (__bf16)0.0f;
  v8f c;
#pragma unroll
  for (int v = 0; v < 8; ++v) c[v] = 0.0f;
  __syncthreads();

  for (int s = 0; s < LL; ++s) {
    const int tt = dir ? (LL - 1 - s) : s;

    // init gate accumulators from precomputed Z (bf16, [t][4U][16] layout)
    v8f acc[4];
#pragma unroll
    for (int g = 0; g < 4; ++g) {
      const __bf16* zp = Z + (size_t)tt * ((size_t)GG * 16)
                           + (size_t)(g * UU + n0 + l15) * 16 + lh8;
      acc[g] = __builtin_convertvector(*(const v8bf*)zp, v8f);
    }

    // h_{t-1} @ U : 16 K-fragments of 32
    for (int kt = 0; kt < 16; ++kt) {
      const __bf16* hp = &hbuf[l15 * 512 + kt * 32 + lh8];
      v16bf a = concat16(*(const v8bf*)hp, *(const v8bf*)(hp + 16));
#pragma unroll
      for (int g = 0; g < 4; ++g) {
        const int nt = g * 32 + w;
        const __bf16* bp = Up + ((size_t)nt * 16 + kt) * 512 + lane * 16;
        v16bf bfr = concat16(*(const v8bf*)bp, *(const v8bf*)(bp + 8));
        acc[g] = __builtin_amdgcn_wmma_f32_16x16x32_bf16(false, a, false, bfr,
                                                         (short)0, acc[g], false, false);
      }
    }

    // fused gate math; i/f/g/o tiles share the (vgpr,lane)->(m,n) mapping
    v8bf hnew;
#pragma unroll
    for (int v = 0; v < 8; ++v) {
      float iv = fsigm(acc[0][v]);
      float fv = fsigm(acc[1][v]);
      float gv = ftanh(acc[2][v]);
      float ov = fsigm(acc[3][v]);
      float cn = fv * c[v] + iv * gv;
      c[v] = cn;
      hnew[v] = (__bf16)(ov * ftanh(cn));
    }

    __syncthreads();  // all waves finished reading h_{t-1}
#pragma unroll
    for (int v = 0; v < 8; ++v) {
      int m = lh8 + v;  // batch row
      hbuf[m * 512 + n0 + l15] = hnew[v];
      H[(size_t)(tt * 16 + m) * ldh + coff + n0 + l15] = hnew[v];
    }
    __syncthreads();  // h_t visible to all waves
  }
}

// ---------------------------------------------------------------------------
extern "C" void kernel_launch(void* const* d_in, const int* in_sizes, int n_in,
                              void* d_out, int out_size, void* d_ws, size_t ws_size,
                              hipStream_t stream) {
  (void)in_sizes; (void)n_in; (void)out_size; (void)ws_size;
  const float* x  = (const float*)d_in[0];
  const float* Wf = (const float*)d_in[1];
  const float* Uf = (const float*)d_in[2];
  const float* bf = (const float*)d_in[3];
  const float* Wb = (const float*)d_in[4];
  const float* Ub = (const float*)d_in[5];
  const float* bb = (const float*)d_in[6];
  const float* Wo = (const float*)d_in[7];
  const float* bo = (const float*)d_in[8];
  float* out = (float*)d_out;
  char* ws = (char*)d_ws;

  const size_t szXbf = (size_t)BB * LL * EE * 2;          // 33.5 MB
  const size_t szW   = (size_t)512 * 2048 * 2;            // 2 MB per matrix
  const size_t szWo  = (size_t)1024 * 512 * 2;            // 1 MB
  const size_t szZ   = (size_t)LL * GG * BB * 2;          // 134 MB each
  const size_t szH1  = (size_t)LL * BB * UU * 2;          // 33.5 MB each
  const size_t szH2  = (size_t)LL * BB * 2 * UU * 2;      // 67 MB

  size_t off = 0;
  __bf16* Xbf  = (__bf16*)(ws + off); off += szXbf;
  __bf16* Wpk  = (__bf16*)(ws + off); off += 8 * szW;     // [Wf0,Wf1,Uf0,Uf1,Wb0,Wb1,Ub0,Ub1]
  __bf16* Wopk = (__bf16*)(ws + off); off += szWo;
  __bf16* Zf   = (__bf16*)(ws + off); off += szZ;
  __bf16* Zb   = (__bf16*)(ws + off); off += szZ;
  __bf16* H1f  = (__bf16*)(ws + off); off += szH1;
  __bf16* H1b  = (__bf16*)(ws + off); off += szH1;
  __bf16* H2   = (__bf16*)(ws + off); off += szH2;

  const size_t wElems = (size_t)512 * 2048;
  __bf16* pWf0 = Wpk + 0 * wElems;
  __bf16* pWf1 = Wpk + 1 * wElems;
  __bf16* pUf0 = Wpk + 2 * wElems;
  __bf16* pUf1 = Wpk + 3 * wElems;
  __bf16* pWb0 = Wpk + 4 * wElems;
  __bf16* pWb1 = Wpk + 5 * wElems;
  __bf16* pUb0 = Wpk + 6 * wElems;
  __bf16* pUb1 = Wpk + 7 * wElems;

  // 1) convert/transpose x, pack all weights to WMMA fragment order
  cvt_x_kernel<<<(BB * LL * EE) / 256, 256, 0, stream>>>(x, Xbf);
  const float* srcs[8] = {Wf, Wf + wElems, Uf, Uf + wElems,
                          Wb, Wb + wElems, Ub, Ub + wElems};
  __bf16* dsts[8] = {pWf0, pWf1, pUf0, pUf1, pWb0, pWb1, pUb0, pUb1};
  for (int i = 0; i < 8; ++i)
    pack_b_kernel<<<256, 256, 0, stream>>>(srcs[i], dsts[i], 512, 2048);
  pack_b_kernel<<<128, 256, 0, stream>>>(Wo, Wopk, 1024, 512);

  dim3 gz(32768 / 128, 2048 / 64);
  // 2) layer 0: input transforms then recurrence (fwd & bwd in parallel blocks)
  gemm_wmma_kernel<<<gz, 256, 0, stream>>>(Xbf, pWf0, bf,        Zf, 32768, 2048, 512, 0);
  gemm_wmma_kernel<<<gz, 256, 0, stream>>>(Xbf, pWb0, bb,        Zb, 32768, 2048, 512, 0);
  lstm_rec_kernel<<<2, 1024, 0, stream>>>(Zf, Zb, pUf0, pUb0, H1f, H1b, 512, 0, 0);

  // 3) layer 1
  gemm_wmma_kernel<<<gz, 256, 0, stream>>>(H1f, pWf1, bf + 2048, Zf, 32768, 2048, 512, 0);
  gemm_wmma_kernel<<<gz, 256, 0, stream>>>(H1b, pWb1, bb + 2048, Zb, 32768, 2048, 512, 0);
  lstm_rec_kernel<<<2, 1024, 0, stream>>>(Zf, Zb, pUf1, pUb1, H2, H2, 1024, 0, 512);

  // 4) output projection: [32768,1024] @ [1024,512] + bo -> f32 [B,L,U]
  dim3 gp(32768 / 128, 512 / 64);
  gemm_wmma_kernel<<<gp, 256, 0, stream>>>(H2, Wopk, bo, out, 32768, 512, 1024, 1);
}
